// MeanPoolStudyHead_87591563035145
// MI455X (gfx1250) — matrix operand order repaired and checked by reading, not compiled, not run on hardware
//
#include <hip/hip_runtime.h>
#include <hip/hip_bf16.h>
#include <stdint.h>

typedef __attribute__((ext_vector_type(16))) _Float16 v16h;
typedef __attribute__((ext_vector_type(8)))  _Float16 v8h;
typedef __attribute__((ext_vector_type(4)))  _Float16 v4h;
typedef __attribute__((ext_vector_type(8)))  float    v8f;
typedef int v4i __attribute__((vector_size(16)));   // 16-byte chunk type for async LDS ops

#define EMB_DIM     1024
#define HIDDEN_DIM  512
#define NUM_CLASSES 14
#define N_INST      262144
#define BATCH       16384
#define LDS_STRIDE  (HIDDEN_DIM + 8)   // 520 f16: breaks LDS bank conflicts (hidden tile)
#define A_STRIDE    (EMB_DIM + 8)      // 1032 f16: row stride 516 dwords, 516%64=4

union Frag16 { v16h v; _Float16 h[16]; uint32_t u[8]; };

// ---- CDNA5 async global->LDS copy (ASYNCcnt path), with safe fallback ------
__device__ __forceinline__ void async_copy_b128(const _Float16* g, _Float16* l) {
#if defined(__gfx1250__) && __has_builtin(__builtin_amdgcn_global_load_async_to_lds_b128)
  __builtin_amdgcn_global_load_async_to_lds_b128(
      (__attribute__((address_space(1))) v4i*)(uintptr_t)g,
      (__attribute__((address_space(3))) v4i*)(uintptr_t)l,
      0, 0);
#else
  *(v8h*)l = *(const v8h*)g;
#endif
}

__device__ __forceinline__ void wait_async0() {
#if defined(__gfx1250__) && __has_builtin(__builtin_amdgcn_s_wait_asynccnt)
  __builtin_amdgcn_s_wait_asynccnt(0);
#elif defined(__gfx1250__)
  asm volatile("s_wait_asynccnt 0x0" ::: "memory");
#endif
}

// ---------------------------------------------------------------------------
// Kernel 1: segment-mean pool (group_idx sorted -> binary search run bounds).
// One block per group; 256 threads x float4 columns = fully coalesced 4KB/row.
// Output: means as f16, row-major [BATCH, EMB_DIM].
// ---------------------------------------------------------------------------
__global__ __launch_bounds__(256) void pool_kernel(
    const float* __restrict__ embs, const int* __restrict__ gidx,
    _Float16* __restrict__ Af16) {
  const int b = blockIdx.x;

  int lo = 0, hi = N_INST;
  while (lo < hi) { int mid = (lo + hi) >> 1; if (gidx[mid] < b) lo = mid + 1; else hi = mid; }
  const int start = lo;
  hi = N_INST;
  while (lo < hi) { int mid = (lo + hi) >> 1; if (gidx[mid] < b + 1) lo = mid + 1; else hi = mid; }
  const int end = lo;

  const int c = threadIdx.x * 4;
  float4 acc = make_float4(0.f, 0.f, 0.f, 0.f);
  for (int r = start; r < end; ++r) {
    if (r + 1 < end)
      __builtin_prefetch(embs + (size_t)(r + 1) * EMB_DIM + c, 0, 1);  // global_prefetch_b8
    const float4 x = *(const float4*)(embs + (size_t)r * EMB_DIM + c);
    acc.x += x.x; acc.y += x.y; acc.z += x.z; acc.w += x.w;
  }
  const float inv = (end > start) ? 1.0f / (float)(end - start) : 0.0f;
  v4h o;
  o[0] = (_Float16)(acc.x * inv); o[1] = (_Float16)(acc.y * inv);
  o[2] = (_Float16)(acc.z * inv); o[3] = (_Float16)(acc.w * inv);
  *(v4h*)(Af16 + (size_t)b * EMB_DIM + c) = o;
}

// ---------------------------------------------------------------------------
// Kernel 2: pack W1 [1024,512] f32 into B-fragment dword order (f16).
// Layout: [K/32][N/16][lane 0..31][dword 0..7]; dword v of lane half h holds
// f16 pair {K=k0, K=k0+1}, k0 = 2v + 8h + (v >= 4 ? 8 : 0) (table order).
// ---------------------------------------------------------------------------
__global__ __launch_bounds__(256) void pack_w1(
    const float* __restrict__ W1, _Float16* __restrict__ W1p) {
  const int i = blockIdx.x * blockDim.x + threadIdx.x;   // dword index
  const int total = (EMB_DIM / 32) * (HIDDEN_DIM / 16) * 32 * 8;
  if (i >= total) return;
  const int v      = i & 7;
  const int lane   = (i >> 3) & 31;
  const int nt     = (i >> 8) % (HIDDEN_DIM / 16);
  const int kk     = (i >> 8) / (HIDDEN_DIM / 16);
  const int laneLo = lane & 15, laneHi = lane >> 4;
  const int kbase  = kk * 32 + 2 * v + (laneHi ? 8 : 0) + (v >= 4 ? 8 : 0);
  const int n      = nt * 16 + laneLo;
  W1p[2 * (size_t)i]     = (_Float16)W1[(size_t)kbase * HIDDEN_DIM + n];
  W1p[2 * (size_t)i + 1] = (_Float16)W1[(size_t)(kbase + 1) * HIDDEN_DIM + n];
}

// Kernel 3: pack W2 [512,14] -> [K/32][lane][dword], N padded 14->16 with 0.
__global__ __launch_bounds__(256) void pack_w2(
    const float* __restrict__ W2, _Float16* __restrict__ W2p) {
  const int i = blockIdx.x * blockDim.x + threadIdx.x;   // dword index
  const int total = (HIDDEN_DIM / 32) * 32 * 8;
  if (i >= total) return;
  const int v      = i & 7;
  const int lane   = (i >> 3) & 31;
  const int kk     = i >> 8;
  const int laneLo = lane & 15, laneHi = lane >> 4;
  const int kbase  = kk * 32 + 2 * v + (laneHi ? 8 : 0) + (v >= 4 ? 8 : 0);
  const int n      = laneLo;
  const float x0 = (n < NUM_CLASSES) ? W2[(size_t)kbase * NUM_CLASSES + n] : 0.f;
  const float x1 = (n < NUM_CLASSES) ? W2[(size_t)(kbase + 1) * NUM_CLASSES + n] : 0.f;
  W2p[2 * (size_t)i]     = (_Float16)x0;
  W2p[2 * (size_t)i + 1] = (_Float16)x1;
}

// ---------------------------------------------------------------------------
// Kernel 4: fused MLP. Block = 16 M-rows, 8 waves; wave w owns 64 N-columns
// (4 WMMA tiles). A-tile (16x1024 f16, 32KB) is staged to LDS once via
// async global->LDS loads (ASYNCcnt), removing the 8x redundant global A
// traffic. Stage1: 32 K-steps of v_wmma_f32_16x16x32_f16 (A from ds_load,
// B from packed global); b1+ReLU -> f16 hidden in LDS. Stage2: wave 0 does
// 16 K-steps vs packed W2, +b2, store 16x14 logits.
// ---------------------------------------------------------------------------
__global__ __launch_bounds__(256) void mlp_kernel(
    const _Float16* __restrict__ A, const _Float16* __restrict__ W1p,
    const float* __restrict__ b1, const _Float16* __restrict__ W2p,
    const float* __restrict__ b2, float* __restrict__ out) {
  __shared__ _Float16 sA[16 * A_STRIDE];     // ~33.0 KB
  __shared__ _Float16 hsm[16 * LDS_STRIDE];  // ~16.6 KB

  const int m0     = blockIdx.x * 16;
  const int wave   = threadIdx.x >> 5;
  const int lane   = threadIdx.x & 31;
  const int laneLo = lane & 15, laneHi = lane >> 4;

  // ---- async stage of the 16x1024 f16 A tile into LDS (2048 x 16B chunks) --
  #pragma unroll
  for (int j = 0; j < (16 * (EMB_DIM / 8)) / 256; ++j) {   // 8 chunks per thread
    const int i   = threadIdx.x + 256 * j;
    const int row = i >> 7;                 // 128 chunks per row
    const int col = (i & 127) * 8;
    async_copy_b128(A + (size_t)(m0 + row) * EMB_DIM + col,
                    sA + row * A_STRIDE + col);
  }
  wait_async0();
  __syncthreads();

  v8f zero;
  #pragma unroll
  for (int r = 0; r < 8; ++r) zero[r] = 0.0f;
  v8f acc[4] = {zero, zero, zero, zero};

  const int nbase = wave * 64;
  const _Float16* arowBase = sA + laneLo * A_STRIDE + (laneHi ? 8 : 0);

  for (int kk = 0; kk < EMB_DIM / 32; ++kk) {
    // A fragment from LDS: interleaved 16-bit A layout (ISA 7.12.2)
    Frag16 a;
    const _Float16* arow = arowBase + kk * 32;
    #pragma unroll
    for (int v = 0; v < 8; ++v) {
      const int koff = (v < 4) ? 2 * v : 2 * v + 8;
      a.u[v] = *(const uint32_t*)(arow + koff);
    }
    #pragma unroll
    for (int t = 0; t < 4; ++t) {
      Frag16 bf;
      const uint32_t* bp = (const uint32_t*)W1p +
          (((size_t)kk * (HIDDEN_DIM / 16) + (nbase >> 4) + t) * 32 + lane) * 8;
      #pragma unroll
      for (int v = 0; v < 8; ++v) bf.u[v] = bp[v];
      acc[t] = __builtin_amdgcn_wmma_f32_16x16x32_f16(
          false, a.v, false, bf.v, (short)0, acc[t], false, false);
    }
  }

  // b1 + ReLU -> f16 hidden tile in LDS (C layout: m = r + 8*laneHi, n = laneLo)
  #pragma unroll
  for (int t = 0; t < 4; ++t) {
    const int n = nbase + t * 16 + laneLo;
    const float bias = b1[n];
    #pragma unroll
    for (int r = 0; r < 8; ++r) {
      const int m = r + (laneHi ? 8 : 0);
      float val = acc[t][r] + bias;
      val = val > 0.f ? val : 0.f;
      hsm[m * LDS_STRIDE + n] = (_Float16)val;
    }
  }
  __syncthreads();

  if (wave == 0) {   // uniform per-wave branch: EXEC all-ones inside (WMMA legal)
    v8f acc2 = zero;
    const _Float16* hrow = hsm + laneLo * LDS_STRIDE + (laneHi ? 8 : 0);
    for (int kk = 0; kk < HIDDEN_DIM / 32; ++kk) {
      Frag16 a;
      #pragma unroll
      for (int v = 0; v < 8; ++v) {
        const int koff = (v < 4) ? 2 * v : 2 * v + 8;
        a.u[v] = *(const uint32_t*)(hrow + kk * 32 + koff);
      }
      Frag16 bf;
      const uint32_t* bp = (const uint32_t*)W2p + ((size_t)kk * 32 + lane) * 8;
      #pragma unroll
      for (int v = 0; v < 8; ++v) bf.u[v] = bp[v];
      acc2 = __builtin_amdgcn_wmma_f32_16x16x32_f16(
          false, a.v, false, bf.v, (short)0, acc2, false, false);
    }
    if (laneLo < NUM_CLASSES) {
      const float bias = b2[laneLo];
      #pragma unroll
      for (int r = 0; r < 8; ++r) {
        const int m = r + (laneHi ? 8 : 0);
        out[(size_t)(m0 + m) * NUM_CLASSES + laneLo] = acc2[r] + bias;
      }
    }
  }
}

// ---------------------------------------------------------------------------
extern "C" void kernel_launch(void* const* d_in, const int* in_sizes, int n_in,
                              void* d_out, int out_size, void* d_ws, size_t ws_size,
                              hipStream_t stream) {
  const float* embs = (const float*)d_in[0];   // inst_embs [262144,1024] f32
  const float* W1   = (const float*)d_in[1];   // [1024,512] f32
  const float* b1   = (const float*)d_in[2];   // [512] f32
  const float* W2   = (const float*)d_in[3];   // [512,14] f32
  const float* b2   = (const float*)d_in[4];   // [14] f32
  const int*   gidx = (const int*)d_in[5];     // [262144] sorted int32
  // d_in[6] = view_idx (unused), d_in[7] = batch_size scalar (compile-time 16384)
  float* out = (float*)d_out;                  // [16384,14] f32

  char* ws = (char*)d_ws;
  _Float16* Af16 = (_Float16*)ws;                                   // 32 MB
  _Float16* W1p  = (_Float16*)(ws + (size_t)BATCH * EMB_DIM * 2);   // 1 MB
  _Float16* W2p  = (_Float16*)((char*)W1p + (size_t)EMB_DIM * HIDDEN_DIM * 2); // 16 KB

  pool_kernel<<<BATCH, 256, 0, stream>>>(embs, gidx, Af16);

  const int w1dw = (EMB_DIM / 32) * (HIDDEN_DIM / 16) * 32 * 8;     // 262144
  pack_w1<<<(w1dw + 255) / 256, 256, 0, stream>>>(W1, W1p);
  const int w2dw = (HIDDEN_DIM / 32) * 32 * 8;                      // 4096
  pack_w2<<<(w2dw + 255) / 256, 256, 0, stream>>>(W2, W2p);

  mlp_kernel<<<BATCH / 16, 256, 0, stream>>>(Af16, W1p, b1, W2p, b2, out);
}
